// CapsuleLayer_45552423142009
// MI455X (gfx1250) — matrix operand order repaired and checked by reading, not compiled, not run on hardware
//
#include <hip/hip_runtime.h>
#include <hip/hip_bf16.h>

// ---------------------------------------------------------------------------
// Capsule layer for MI455X (gfx1250, wave32, WMMA, async LDS-DMA).
//
// Roofline: dominant GEMM [131072x768]x[768x128] = 25.8 GFLOP over ~470MB of
// HBM traffic (x fp32 = 402MB -> ~18us at 23.3 TB/s). bf16 WMMA (16x16x32,
// fp32 accum) is required to stay memory-bound; fp32 WMMA (K=4) has 8x lower
// matrix throughput and would dominate. x is staged to LDS with
// global_load_async_to_lds_b128 (ASYNCcnt), double-buffered so the next
// k-chunk's DMA overlaps the current chunk's WMMAs.
//
// ws layout (bytes):
//   [0,          196608)  Wp bf16, swizzled into WMMA B-fragments (192KB)
//   [196608,     237568)  W  bf16, swizzled B-fragments, K padded 16->32 (40KB)
//   [237568,  +33554432)  u  bf16 [B][128] intermediate (32MB)
// ---------------------------------------------------------------------------

typedef __attribute__((ext_vector_type(16))) __bf16 v16bf;
typedef __attribute__((ext_vector_type(8)))  float  v8f;
typedef __attribute__((ext_vector_type(8)))  int    v8i;
typedef __attribute__((ext_vector_type(4)))  int    v4i;

// Builtin expects: arg0 = int4-vector pointer in device/global AS(1) (non-const),
// arg1 = LDS AS(3) pointer (per round-2 diagnostic).
typedef __attribute__((address_space(1))) v4i* gv4i_p;
typedef __attribute__((address_space(3))) v4i* lv4i_p;

#define B_TOTAL 131072
#define D_IN    768
#define NP      8
#define PD      16
#define NC      5
#define CDIM    16
#define KT      (D_IN / 32)   // 24 k-chunks of 32
#define NT      NP            // 8 n-tiles of 16 (N = 128)

#define WS_WP_BYTES (KT * NT * 32 * 32)     // 196608
#define WS_W_OFF    WS_WP_BYTES
#define WS_W_BYTES  (NP * NC * 32 * 32)     // 40960
#define WS_U_OFF    (WS_W_OFF + WS_W_BYTES) // 237568 (16B aligned)

__device__ __forceinline__ unsigned short f2bf(float f) {
  unsigned int u = __builtin_bit_cast(unsigned int, f);
  u += 0x7fffu + ((u >> 16) & 1u);            // round-to-nearest-even
  return (unsigned short)(u >> 16);
}
__device__ __forceinline__ unsigned int packbf(float a, float b) {
  return (unsigned int)f2bf(a) | ((unsigned int)f2bf(b) << 16);
}
__device__ __forceinline__ float bf_lo(unsigned int d) {
  return __builtin_bit_cast(float, d << 16);
}
__device__ __forceinline__ float bf_hi(unsigned int d) {
  return __builtin_bit_cast(float, d & 0xffff0000u);
}
__device__ __forceinline__ float squash_scale(float sq) {
  return (sq / (1.0f + sq)) / (sqrtf(sq) + 1e-8f);
}

// ---------------------------------------------------------------------------
// K0: convert weights fp32 -> bf16 in fragment-ready swizzled layouts.
// B-fragment layout assumption (32x16 bf16): lane l: n = l&15, h = l>>4;
// dword v<4: K = 8h + 2v; v>=4: K = 16 + 8h + 2(v-4) (+1 in high half of dword).
// ---------------------------------------------------------------------------
__global__ void caps_convert(const float* __restrict__ Wp,
                             const float* __restrict__ W,
                             unsigned int* __restrict__ wp_swz,
                             unsigned int* __restrict__ w_swz) {
  const int wp_dwords = KT * NT * 32 * 8;   // 49152
  const int w_dwords  = NP * NC * 32 * 8;   // 10240
  int idx = blockIdx.x * blockDim.x + threadIdx.x;
  if (idx < wp_dwords) {
    int v = idx & 7, lane = (idx >> 3) & 31, nt = (idx >> 8) & 7, kt = idx >> 11;
    int h = lane >> 4, o = lane & 15;
    int k0 = kt * 32 + (v < 4 ? 8 * h + 2 * v : 16 + 8 * h + 2 * (v - 4));
    // Wp[p][i][o], GEMM n = p*16+o  => p == nt, K == i
    float f0 = Wp[(nt * D_IN + k0) * PD + o];
    float f1 = Wp[(nt * D_IN + k0 + 1) * PD + o];
    wp_swz[idx] = packbf(f0, f1);
  } else if (idx < wp_dwords + w_dwords) {
    int i2 = idx - wp_dwords;
    int v = i2 & 7, lane = (i2 >> 3) & 31, frag = i2 >> 8;
    int p = frag / NC, ct = frag % NC;
    int h = lane >> 4, j = lane & 15;
    unsigned int d = 0u;                    // K >= 16 is zero padding
    if (v < 4) {
      int o0 = 8 * h + 2 * v;
      float f0 = W[((p * NC + ct) * PD + o0) * CDIM + j];
      float f1 = W[((p * NC + ct) * PD + o0 + 1) * CDIM + j];
      d = packbf(f0, f1);
    }
    w_swz[i2] = d;
  }
}

// ---------------------------------------------------------------------------
// K1: primary capsules.  u = squash(x @ Wp + bp), output bf16 [B][128].
// 256 threads = 8 waves; wave owns m-tile (16 rows) x all 8 n-tiles.
// x staged fp32 to LDS via global_load_async_to_lds_b128 (double buffered);
// converted to bf16 while building A fragments.
// A 16-bit fragment (ISA layout): lane l: M = l&15, h = l>>4;
//   VGPR v<4:  K = 8h + 2v(+1);  VGPR v>=4: K = 16 + 8h + 2(v-4)(+1).
// C fragment: lane l: N = l&15; VGPR r: M = r + 8h.
// ---------------------------------------------------------------------------
__device__ __forceinline__ void issue_x_chunk(const float* x, float* dst_lds,
                                              int bbase, int tid, int kt) {
  int row = tid & 127, kh = tid >> 7;       // 2 threads per row, 16 floats each
  gv4i_p g = (gv4i_p)(x + (size_t)(bbase + row) * D_IN + kt * 32 + kh * 16);
  lv4i_p l = (lv4i_p)(dst_lds + row * 32 + kh * 16);
  // IOFFSET is added to both the global and LDS address (ISA 15.18.3)
  __builtin_amdgcn_global_load_async_to_lds_b128(g, l, 0, 0);
  __builtin_amdgcn_global_load_async_to_lds_b128(g, l, 16, 0);
  __builtin_amdgcn_global_load_async_to_lds_b128(g, l, 32, 0);
  __builtin_amdgcn_global_load_async_to_lds_b128(g, l, 48, 0);
}

__global__ void __launch_bounds__(256)
caps_primary(const float* __restrict__ x,
             const float* __restrict__ bp,
             const unsigned int* __restrict__ wp_swz,
             unsigned short* __restrict__ u_out) {
  __shared__ __align__(16) float xs[2][128 * 32];   // 2 x 16KB fp32 tiles
  const int tid = threadIdx.x;
  const int lane = tid & 31, wave = tid >> 5;
  const int bbase = blockIdx.x * 128;
  const int h = lane >> 4, m = lane & 15, o = lane & 15;

  v8f acc[NT] = {};

  issue_x_chunk(x, xs[0], bbase, tid, 0);           // prologue DMA

  for (int kt = 0; kt < KT; ++kt) {
    const int buf = kt & 1;
    if (kt + 1 < KT) {
      issue_x_chunk(x, xs[buf ^ 1], bbase, tid, kt + 1);  // overlap next DMA
      // 4 newer async issues outstanding are OK; chunk kt's 4 must be done.
      asm volatile("s_wait_asynccnt 0x4" ::: "memory");
    } else {
      asm volatile("s_wait_asynccnt 0x0" ::: "memory");
    }
    __syncthreads();   // make every thread's DMA'd rows visible to all waves

    // A fragment: read 16 fp32 from LDS, convert to bf16 pairs
    const float* arow = &xs[buf][(wave * 16 + m) * 32];
    float4 a0 = *(const float4*)(arow + 8 * h);
    float4 a1 = *(const float4*)(arow + 8 * h + 4);
    float4 a2 = *(const float4*)(arow + 16 + 8 * h);
    float4 a3 = *(const float4*)(arow + 16 + 8 * h + 4);
    v8i ar = {(int)packbf(a0.x, a0.y), (int)packbf(a0.z, a0.w),
              (int)packbf(a1.x, a1.y), (int)packbf(a1.z, a1.w),
              (int)packbf(a2.x, a2.y), (int)packbf(a2.z, a2.w),
              (int)packbf(a3.x, a3.y), (int)packbf(a3.z, a3.w)};
    v16bf a = __builtin_bit_cast(v16bf, ar);

#pragma unroll
    for (int nt = 0; nt < NT; ++nt) {
      const int4* wb =
          (const int4*)(wp_swz + ((size_t)((kt * NT + nt) * 32 + lane)) * 8);
      int4 b0 = wb[0], b1 = wb[1];
      v8i br = {b0.x, b0.y, b0.z, b0.w, b1.x, b1.y, b1.z, b1.w};
      v16bf bm = __builtin_bit_cast(v16bf, br);
      acc[nt] = __builtin_amdgcn_wmma_f32_16x16x32_bf16(
          false, a, false, bm, (short)0, acc[nt], false, false);
    }
    __syncthreads();   // buf will be DMA-overwritten at iteration kt+1 (chunk kt+2)
  }

  // epilogue: bias + squash over PD (16 lanes of a half) + store bf16
#pragma unroll
  for (int nt = 0; nt < NT; ++nt) {
    float bias = bp[nt * PD + o];
#pragma unroll
    for (int r = 0; r < 8; ++r) {
      float val = acc[nt][r] + bias;
      float sq = val * val;
      sq += __shfl_xor(sq, 1);
      sq += __shfl_xor(sq, 2);
      sq += __shfl_xor(sq, 4);
      sq += __shfl_xor(sq, 8);        // sum over the 16-lane half (fixed row)
      val *= squash_scale(sq);
      int row = bbase + wave * 16 + r + 8 * h;
      u_out[(size_t)row * 128 + nt * 16 + o] = f2bf(val);
    }
  }
}

// ---------------------------------------------------------------------------
// K2: votes (WMMA, K=16 padded to 32) + 3 dynamic-routing iterations.
// 64 threads = 2 waves; wave owns 16 rows; u_hat staged in LDS as bf16
// (16 rows x 8p x 5c x 16j = 20KB per wave).
// Routing lane map: row m = lane&15, j-half = lane>>4 (8 j's per lane);
// cross-half reductions via __shfl_xor(.,16).
// ---------------------------------------------------------------------------
__global__ void __launch_bounds__(64)
caps_route(const unsigned short* __restrict__ u_in,
           const unsigned int* __restrict__ w_swz,
           float* __restrict__ out) {
  __shared__ __align__(16) unsigned short uh[2][16 * NP * NC * CDIM];  // 40KB
  const int tid = threadIdx.x, lane = tid & 31, w = tid >> 5;
  const int bbase = blockIdx.x * 32 + w * 16;
  const int h = lane >> 4, m = lane & 15;

  // ---- votes: u_hat[m][p][c][j] = sum_o u[m][p][o] * W[p][c][o][j] ----
#pragma unroll
  for (int p = 0; p < NP; ++p) {
    // A fragment: real K = o in [0,16); VGPRs 4..7 are zero padding
    const int4* ua =
        (const int4*)(u_in + (size_t)(bbase + m) * 128 + p * 16 + 8 * h);
    int4 alo = *ua;
    v8i ar = {alo.x, alo.y, alo.z, alo.w, 0, 0, 0, 0};
    v16bf a = __builtin_bit_cast(v16bf, ar);
#pragma unroll
    for (int ct = 0; ct < NC; ++ct) {
      const int4* wb =
          (const int4*)(w_swz + ((size_t)((p * NC + ct) * 32 + lane)) * 8);
      int4 b0 = wb[0], b1 = wb[1];
      v8i br = {b0.x, b0.y, b0.z, b0.w, b1.x, b1.y, b1.z, b1.w};
      v16bf bm = __builtin_bit_cast(v16bf, br);
      v8f c = {};
      c = __builtin_amdgcn_wmma_f32_16x16x32_bf16(
          false, a, false, bm, (short)0, c, false, false);
      int j = lane & 15;
#pragma unroll
      for (int r = 0; r < 8; ++r) {
        int mr = r + 8 * h;
        uh[w][((mr * NP + p) * NC + ct) * CDIM + j] = f2bf(c[r]);
      }
    }
  }
  __syncthreads();

  // ---- routing ----
  const unsigned short* myuh = &uh[w][(m * NP) * NC * CDIM];
  float bl[NP * NC];
#pragma unroll
  for (int i = 0; i < NP * NC; ++i) bl[i] = 0.0f;
  float sv[NC * 8];

#pragma unroll 1
  for (int it = 0; it < 3; ++it) {
#pragma unroll
    for (int i = 0; i < NC * 8; ++i) sv[i] = 0.0f;

#pragma unroll
    for (int p = 0; p < NP; ++p) {
      // softmax over C=5 for this (row, p)
      float mx = bl[p * NC];
#pragma unroll
      for (int c = 1; c < NC; ++c) mx = fmaxf(mx, bl[p * NC + c]);
      float e[NC], sum = 0.0f;
#pragma unroll
      for (int c = 0; c < NC; ++c) { e[c] = __expf(bl[p * NC + c] - mx); sum += e[c]; }
      float inv = 1.0f / sum;
#pragma unroll
      for (int c = 0; c < NC; ++c) {
        float cc = e[c] * inv;
        unsigned int d0, d1, d2, d3;
        {
          int4 d = *(const int4*)&myuh[(p * NC + c) * CDIM + 8 * h];
          d0 = (unsigned int)d.x; d1 = (unsigned int)d.y;
          d2 = (unsigned int)d.z; d3 = (unsigned int)d.w;
        }
        sv[c * 8 + 0] += cc * bf_lo(d0); sv[c * 8 + 1] += cc * bf_hi(d0);
        sv[c * 8 + 2] += cc * bf_lo(d1); sv[c * 8 + 3] += cc * bf_hi(d1);
        sv[c * 8 + 4] += cc * bf_lo(d2); sv[c * 8 + 5] += cc * bf_hi(d2);
        sv[c * 8 + 6] += cc * bf_lo(d3); sv[c * 8 + 7] += cc * bf_hi(d3);
      }
    }

    // squash over CD=16 (own 8 + partner half)
#pragma unroll
    for (int c = 0; c < NC; ++c) {
      float sq = 0.0f;
#pragma unroll
      for (int t = 0; t < 8; ++t) sq += sv[c * 8 + t] * sv[c * 8 + t];
      sq += __shfl_xor(sq, 16);
      float sc = squash_scale(sq);
#pragma unroll
      for (int t = 0; t < 8; ++t) sv[c * 8 + t] *= sc;
    }

    if (it < 2) {  // agreement: b += sum_j u_hat * v
#pragma unroll
      for (int p = 0; p < NP; ++p) {
#pragma unroll
        for (int c = 0; c < NC; ++c) {
          unsigned int d0, d1, d2, d3;
          {
            int4 d = *(const int4*)&myuh[(p * NC + c) * CDIM + 8 * h];
            d0 = (unsigned int)d.x; d1 = (unsigned int)d.y;
            d2 = (unsigned int)d.z; d3 = (unsigned int)d.w;
          }
          float part = bf_lo(d0) * sv[c * 8 + 0] + bf_hi(d0) * sv[c * 8 + 1] +
                       bf_lo(d1) * sv[c * 8 + 2] + bf_hi(d1) * sv[c * 8 + 3] +
                       bf_lo(d2) * sv[c * 8 + 4] + bf_hi(d2) * sv[c * 8 + 5] +
                       bf_lo(d3) * sv[c * 8 + 6] + bf_hi(d3) * sv[c * 8 + 7];
          part += __shfl_xor(part, 16);
          bl[p * NC + c] += part;
        }
      }
    }
  }

  // ---- output v [B][C][CD] fp32 ----
  float* op = out + (size_t)(bbase + m) * NC * CDIM;
#pragma unroll
  for (int c = 0; c < NC; ++c) {
    float4 s0 = {sv[c * 8 + 0], sv[c * 8 + 1], sv[c * 8 + 2], sv[c * 8 + 3]};
    float4 s1 = {sv[c * 8 + 4], sv[c * 8 + 5], sv[c * 8 + 6], sv[c * 8 + 7]};
    *(float4*)(op + c * CDIM + 8 * h) = s0;
    *(float4*)(op + c * CDIM + 8 * h + 4) = s1;
  }
}

// ---------------------------------------------------------------------------
extern "C" void kernel_launch(void* const* d_in, const int* in_sizes, int n_in,
                              void* d_out, int out_size, void* d_ws, size_t ws_size,
                              hipStream_t stream) {
  const float* x  = (const float*)d_in[0];   // [B, 768]
  const float* Wp = (const float*)d_in[1];   // [8, 768, 16]
  const float* bp = (const float*)d_in[2];   // [8, 16]
  const float* W  = (const float*)d_in[3];   // [8, 5, 16, 16]
  float* out = (float*)d_out;                // [B, 5, 16]

  char* ws = (char*)d_ws;                    // needs ~33.8 MB
  unsigned int*   wp_swz = (unsigned int*)(ws);
  unsigned int*   w_swz  = (unsigned int*)(ws + WS_W_OFF);
  unsigned short* u_bf16 = (unsigned short*)(ws + WS_U_OFF);

  const int conv_elems = KT * NT * 32 * 8 + NP * NC * 32 * 8;  // 59392
  caps_convert<<<(conv_elems + 255) / 256, 256, 0, stream>>>(Wp, W, wp_swz, w_swz);
  caps_primary<<<B_TOTAL / 128, 256, 0, stream>>>(x, bp, wp_swz, u_bf16);
  caps_route<<<B_TOTAL / 32, 64, 0, stream>>>(u_bf16, w_swz, out);
}
